// SwinTransformerBlock_24086176596574
// MI455X (gfx1250) — matrix operand order repaired and checked by reading, not compiled, not run on hardware
//
#include <hip/hip_runtime.h>
#include <hip/hip_bf16.h>

// ---------------- types / helpers ----------------
typedef __bf16 bf16;
typedef __attribute__((ext_vector_type(8)))  __bf16 bf16x8;
typedef __attribute__((ext_vector_type(16))) __bf16 bf16x16;
typedef __attribute__((ext_vector_type(8)))  float  f32x8;

union FragA { bf16x16 v; bf16x8 h[2]; };

__device__ __forceinline__ bf16 tobf(float f) {
    unsigned int u = __builtin_bit_cast(unsigned int, f);
    unsigned int r = (u + 0x7fffu + ((u >> 16) & 1u)) >> 16;
    unsigned short s = (unsigned short)r;
    return __builtin_bit_cast(bf16, s);
}

// issue one per-lane 16B async global->LDS copy (CDNA5, tracked by ASYNCcnt)
__device__ __forceinline__ void async_g2l_b128(unsigned ldsOff, const void* gp) {
    asm volatile("global_load_async_to_lds_b128 %0, %1, off"
                 :: "v"(ldsOff), "v"(gp) : "memory");
}
__device__ __forceinline__ void wait_async0() {
    asm volatile("s_wait_asynccnt 0x0" ::: "memory");
}

// problem constants
#define BB   16
#define HH   224
#define WW   224
#define CC   96
#define NHD  3
#define WS7  7
#define SS3  3
#define NTOK 49            // tokens per window
#define NWIN 1024          // windows per image
#define LL   (HH*WW)       // 50176
#define MTOT (BB*LL)       // 802816 rows
#define HID  384

// ---------------- weight conversion ----------------
__global__ void f2bf_kernel(const float* __restrict__ in, bf16* __restrict__ out, int n) {
    int i = blockIdx.x * 256 + threadIdx.x;
    if (i < n) out[i] = tobf(in[i]);
}

// ---------------- LayerNorm (+ optional shift/window remap), f32 -> bf16 ----------------
// one wave (32 lanes) per token, C=96 -> 3 channels/lane
__global__ __launch_bounds__(256) void ln_kernel(
    const float* __restrict__ x, const float* __restrict__ g, const float* __restrict__ b,
    bf16* __restrict__ out, int remap)
{
    int m = blockIdx.x * 8 + (threadIdx.x >> 5);
    int lane = threadIdx.x & 31;
    size_t t;
    if (remap) {
        int bimg = m / LL, r = m % LL;
        int wi = r / NTOK, n = r % NTOK;
        int gy = ((wi >> 5) * WS7 + n / WS7 + SS3) % HH;
        int gx = ((wi & 31) * WS7 + n % WS7 + SS3) % WW;
        t = (size_t)bimg * LL + (size_t)gy * WW + gx;
    } else {
        t = (size_t)m;
    }
    float v0 = x[t * CC + lane];
    float v1 = x[t * CC + 32 + lane];
    float v2 = x[t * CC + 64 + lane];
    float s  = v0 + v1 + v2;
    float sq = v0 * v0 + v1 * v1 + v2 * v2;
    #pragma unroll
    for (int o = 16; o > 0; o >>= 1) {
        s  += __shfl_xor(s,  o, 32);
        sq += __shfl_xor(sq, o, 32);
    }
    float mean = s * (1.0f / 96.0f);
    float inv  = rsqrtf(sq * (1.0f / 96.0f) - mean * mean + 1e-5f);
    out[(size_t)m * CC + lane]      = tobf((v0 - mean) * inv * g[lane]      + b[lane]);
    out[(size_t)m * CC + 32 + lane] = tobf((v1 - mean) * inv * g[lane + 32] + b[lane + 32]);
    out[(size_t)m * CC + 64 + lane] = tobf((v2 - mean) * inv * g[lane + 64] + b[lane + 64]);
}

// ---------------- generic tiled WMMA GEMM ----------------
// C[M,Nc] = A[M,K](bf16 row-major) @ W[Nc,K]^T (bf16 row-major) + bias, mode epilogue.
// Block tile 128x32, 256 threads = 8 waves. Wave w owns row strip w*16..w*16+15 and
// BOTH 16-col B tiles (A fragment reused by 2 WMMAs). K staged to LDS with
// GLOBAL_LOAD_ASYNC_TO_LDS_B128 in 96-wide chunks.
#define KC 96
#define BM 128
__global__ __launch_bounds__(256) void gemm_wmma_kernel(
    const bf16* __restrict__ A, const bf16* __restrict__ W, const float* __restrict__ bias,
    bf16* __restrict__ outB, float* __restrict__ outF, const float* __restrict__ res,
    int M, int K, int Nc, int mode)
{
    __shared__ __align__(16) bf16 As[BM * KC];   // 24 KB
    __shared__ __align__(16) bf16 Bs[32 * KC];   // 6 KB
    const int tid  = threadIdx.x;
    const int lane = tid & 31, wave = tid >> 5;
    const int hi   = lane >> 4, l16 = lane & 15;
    const int rowBase = blockIdx.x * BM;
    const int colBase = blockIdx.y * 32;

    // LDS byte offsets (aperture rule: LDS offset = low 32 bits of generic address)
    const unsigned asBase = (unsigned)(uintptr_t)&As[0];
    const unsigned bsBase = (unsigned)(uintptr_t)&Bs[0];

    f32x8 acc0 = {}, acc1 = {};
    for (int kc = 0; kc < K; kc += KC) {
        // async stage-in: 16B per lane per issue, no VGPR round-trip
        for (int cix = tid; cix < BM * 12; cix += 256) {     // 128 rows * 12 chunks
            int r = cix / 12, c8 = cix % 12;
            async_g2l_b128(asBase + (unsigned)(r * KC + c8 * 8) * 2u,
                           &A[(size_t)(rowBase + r) * K + kc + c8 * 8]);
        }
        for (int cix = tid; cix < 32 * 12; cix += 256) {     // 32 rows * 12 chunks
            int r = cix / 12, c8 = cix % 12;
            async_g2l_b128(bsBase + (unsigned)(r * KC + c8 * 8) * 2u,
                           &W[(size_t)(colBase + r) * K + kc + c8 * 8]);
        }
        if (kc + KC < K)  // CDNA5 global_prefetch of next K-chunk
            __builtin_prefetch(&A[(size_t)(rowBase + (tid >> 1)) * K + kc + KC], 0, 0);
        wait_async0();
        __syncthreads();

        const int ar = wave * 16 + l16;
        #pragma unroll
        for (int ks = 0; ks < 3; ++ks) {
            const int k0 = ks * 32;
            FragA af, b0, b1;
            af.h[0] = *(const bf16x8*)&As[ar * KC + k0 + hi * 16];
            af.h[1] = *(const bf16x8*)&As[ar * KC + k0 + hi * 16 + 8];
            b0.h[0] = *(const bf16x8*)&Bs[l16 * KC + k0 + hi * 16];
            b0.h[1] = *(const bf16x8*)&Bs[l16 * KC + k0 + hi * 16 + 8];
            b1.h[0] = *(const bf16x8*)&Bs[(16 + l16) * KC + k0 + hi * 16];
            b1.h[1] = *(const bf16x8*)&Bs[(16 + l16) * KC + k0 + hi * 16 + 8];
            acc0 = __builtin_amdgcn_wmma_f32_16x16x32_bf16(
                false, af.v, false, b0.v, (short)0, acc0, false, false);
            acc1 = __builtin_amdgcn_wmma_f32_16x16x32_bf16(
                false, af.v, false, b1.v, (short)0, acc1, false, false);
        }
        __syncthreads();
    }

    const int row0 = rowBase + wave * 16;
    f32x8 accs[2] = {acc0, acc1};
    #pragma unroll
    for (int nt = 0; nt < 2; ++nt) {
        const int col = colBase + nt * 16 + l16;
        const float bv = bias[col];
        #pragma unroll
        for (int v = 0; v < 8; ++v) {
            int row = row0 + v + 8 * hi;
            float val = accs[nt][v] + bv;
            if (mode == 0) {                       // QKV: scale q (cols 0..95) incl. bias
                if (col < 96) val *= 0.17677669529663687f;   // 32^-0.5
                outB[(size_t)row * Nc + col] = tobf(val);
            } else if (mode == 1) {                // proj: window-reverse + unshift + residual
                int bimg = row / LL, r = row % LL;
                int wi = r / NTOK, n = r % NTOK;
                int gy = ((wi >> 5) * WS7 + n / WS7 + SS3) % HH;
                int gx = ((wi & 31) * WS7 + n % WS7 + SS3) % WW;
                size_t t = (size_t)bimg * LL + (size_t)gy * WW + gx;
                outF[t * CC + col] = res[t * CC + col] + val;
            } else if (mode == 2) {                // FC1 + exact GELU
                val = 0.5f * val * (1.0f + erff(val * 0.70710678118654752f));
                outB[(size_t)row * Nc + col] = tobf(val);
            } else {                               // FC2 + residual -> final out
                outF[(size_t)row * Nc + col] = res[(size_t)row * Nc + col] + val;
            }
        }
    }
}

// ---------------- windowed attention (one block per window*head, 4 waves) ----------------
__global__ __launch_bounds__(128) void attn_kernel(
    const bf16* __restrict__ qkv, const float* __restrict__ rpb, bf16* __restrict__ out)
{
    __shared__ __align__(16) bf16 q_s[64 * 32];
    __shared__ __align__(16) bf16 k_s[64 * 32];
    __shared__ __align__(16) bf16 vT_s[32 * 64];
    __shared__ __align__(16) float S_s[64 * 64];
    __shared__ __align__(16) bf16 p_s[64 * 64];

    const int tid  = threadIdx.x;
    const int lane = tid & 31, wave = tid >> 5;
    const int hi   = lane >> 4, l16 = lane & 15;
    const int head = blockIdx.x % NHD;
    const int win  = blockIdx.x / NHD;
    const size_t base = (size_t)win * NTOK * (3 * CC) + (size_t)head * 32;

    // stage q, k, v (padded to 64 tokens; v transposed for B-fragment K-contiguity)
    for (int idx = tid; idx < 2048; idx += 128) {
        int t = idx >> 5, d = idx & 31;
        bf16 qv = tobf(0.f), kv = tobf(0.f), vv = tobf(0.f);
        if (t < NTOK) {
            size_t p = base + (size_t)t * (3 * CC);
            qv = qkv[p + d];
            kv = qkv[p + 96 + d];
            vv = qkv[p + 192 + d];
        }
        q_s[idx] = qv; k_s[idx] = kv; vT_s[d * 64 + t] = vv;
    }
    __syncthreads();

    // phase 1: S = q @ k^T  (+ rel-pos bias + shift mask)
    {
        const int mt = wave;
        const int ar = mt * 16 + l16;
        FragA af;
        af.h[0] = *(const bf16x8*)&q_s[ar * 32 + hi * 16];
        af.h[1] = *(const bf16x8*)&q_s[ar * 32 + hi * 16 + 8];
        const int wi  = win % NWIN;
        const int wy0 = (wi >> 5) * WS7, wx0 = (wi & 31) * WS7;
        for (int nt = 0; nt < 4; ++nt) {
            FragA bfr;
            const int br = nt * 16 + l16;
            bfr.h[0] = *(const bf16x8*)&k_s[br * 32 + hi * 16];
            bfr.h[1] = *(const bf16x8*)&k_s[br * 32 + hi * 16 + 8];
            f32x8 acc = {};
            acc = __builtin_amdgcn_wmma_f32_16x16x32_bf16(
                false, af.v, false, bfr.v, (short)0, acc, false, false);
            #pragma unroll
            for (int v = 0; v < 8; ++v) {
                int i = mt * 16 + v + 8 * hi;
                int j = nt * 16 + l16;
                float sv = -1e30f;
                if (i < NTOK && j < NTOK) {
                    int yi = i / WS7, xi = i % WS7, yj = j / WS7, xj = j % WS7;
                    int ridx = (yi - yj + WS7 - 1) * (2 * WS7 - 1) + (xi - xj + WS7 - 1);
                    float bb = rpb[ridx * NHD + head];
                    int gyi = wy0 + yi, gxi = wx0 + xi, gyj = wy0 + yj, gxj = wx0 + xj;
                    int ri = (gyi < HH - WS7 ? 0 : (gyi < HH - SS3 ? 1 : 2)) * 3
                           + (gxi < WW - WS7 ? 0 : (gxi < WW - SS3 ? 1 : 2));
                    int rj = (gyj < HH - WS7 ? 0 : (gyj < HH - SS3 ? 1 : 2)) * 3
                           + (gxj < WW - WS7 ? 0 : (gxj < WW - SS3 ? 1 : 2));
                    sv = acc[v] + bb + (ri == rj ? 0.f : -100.f);
                }
                S_s[i * 64 + j] = sv;
            }
        }
    }
    __syncthreads();

    // softmax over each valid row (cols 0..48), P padded with zeros to 64x64
    if (tid < 64) {
        int r = tid;
        if (r < NTOK) {
            float mx = -1e30f;
            for (int j = 0; j < NTOK; ++j) mx = fmaxf(mx, S_s[r * 64 + j]);
            float sum = 0.f;
            for (int j = 0; j < NTOK; ++j) {
                float e = expf(S_s[r * 64 + j] - mx);
                S_s[r * 64 + j] = e;
                sum += e;
            }
            float rs = 1.0f / sum;
            for (int j = 0; j < NTOK; ++j) p_s[r * 64 + j] = tobf(S_s[r * 64 + j] * rs);
            for (int j = NTOK; j < 64; ++j) p_s[r * 64 + j] = tobf(0.f);
        } else {
            for (int j = 0; j < 64; ++j) p_s[r * 64 + j] = tobf(0.f);
        }
    }
    __syncthreads();

    // phase 2: O = P @ V
    {
        const int mt = wave;
        const int ar = mt * 16 + l16;
        for (int nt = 0; nt < 2; ++nt) {
            f32x8 acc = {};
            #pragma unroll
            for (int ks = 0; ks < 2; ++ks) {
                const int k0 = ks * 32;
                FragA af, bfr;
                af.h[0]  = *(const bf16x8*)&p_s[ar * 64 + k0 + hi * 16];
                af.h[1]  = *(const bf16x8*)&p_s[ar * 64 + k0 + hi * 16 + 8];
                const int bn = nt * 16 + l16;
                bfr.h[0] = *(const bf16x8*)&vT_s[bn * 64 + k0 + hi * 16];
                bfr.h[1] = *(const bf16x8*)&vT_s[bn * 64 + k0 + hi * 16 + 8];
                acc = __builtin_amdgcn_wmma_f32_16x16x32_bf16(
                    false, af.v, false, bfr.v, (short)0, acc, false, false);
            }
            #pragma unroll
            for (int v = 0; v < 8; ++v) {
                int i = mt * 16 + v + 8 * hi;
                int d = nt * 16 + l16;
                if (i < NTOK)
                    out[((size_t)win * NTOK + i) * CC + head * 32 + d] = tobf(acc[v]);
            }
        }
    }
}

// ---------------- host-side orchestration ----------------
extern "C" void kernel_launch(void* const* d_in, const int* in_sizes, int n_in,
                              void* d_out, int out_size, void* d_ws, size_t ws_size,
                              hipStream_t stream) {
    const float* x      = (const float*)d_in[0];
    const float* qkv_w  = (const float*)d_in[1];
    const float* qkv_b  = (const float*)d_in[2];
    const float* proj_w = (const float*)d_in[3];
    const float* proj_b = (const float*)d_in[4];
    const float* rpb    = (const float*)d_in[5];
    const float* n1g    = (const float*)d_in[6];
    const float* n1b    = (const float*)d_in[7];
    const float* n2g    = (const float*)d_in[8];
    const float* n2b    = (const float*)d_in[9];
    const float* fc1_w  = (const float*)d_in[10];
    const float* fc1_b  = (const float*)d_in[11];
    const float* fc2_w  = (const float*)d_in[12];
    const float* fc2_b  = (const float*)d_in[13];

    char* ws = (char*)d_ws;
    size_t off = 0;
    bf16*  ywin    = (bf16*)(ws + off); off += (size_t)MTOT * CC * 2;        // 154 MB
    bf16*  qkvbuf  = (bf16*)(ws + off); off += (size_t)MTOT * 3 * CC * 2;    // 462 MB
    bf16*  attnout = (bf16*)(ws + off); off += (size_t)MTOT * CC * 2;        // 154 MB
    float* x2      = (float*)(ws + off); off += (size_t)MTOT * CC * 4;       // 308 MB
    bf16*  zbuf    = (bf16*)(ws + off); off += (size_t)MTOT * CC * 2;        // 154 MB
    bf16*  h1      = (bf16*)(ws + off); off += (size_t)MTOT * HID * 2;       // 617 MB
    bf16*  qkv_wb  = (bf16*)(ws + off); off += (size_t)3 * CC * CC * 2;
    bf16*  proj_wb = (bf16*)(ws + off); off += (size_t)CC * CC * 2;
    bf16*  fc1_wb  = (bf16*)(ws + off); off += (size_t)HID * CC * 2;
    bf16*  fc2_wb  = (bf16*)(ws + off); off += (size_t)CC * HID * 2;

    // weight conversion to bf16
    f2bf_kernel<<<(3*CC*CC + 255)/256, 256, 0, stream>>>(qkv_w, qkv_wb, 3*CC*CC);
    f2bf_kernel<<<(CC*CC   + 255)/256, 256, 0, stream>>>(proj_w, proj_wb, CC*CC);
    f2bf_kernel<<<(HID*CC  + 255)/256, 256, 0, stream>>>(fc1_w, fc1_wb, HID*CC);
    f2bf_kernel<<<(CC*HID  + 255)/256, 256, 0, stream>>>(fc2_w, fc2_wb, CC*HID);

    // LN1 + cyclic shift + window partition -> bf16 (window order)
    ln_kernel<<<MTOT/8, 256, 0, stream>>>(x, n1g, n1b, ywin, 1);

    // QKV projection (mode 0)
    gemm_wmma_kernel<<<dim3(MTOT/BM, (3*CC)/32), 256, 0, stream>>>(
        ywin, qkv_wb, qkv_b, qkvbuf, nullptr, nullptr, MTOT, CC, 3*CC, 0);

    // windowed attention
    attn_kernel<<<(BB * NWIN) * NHD, 128, 0, stream>>>(qkvbuf, rpb, attnout);

    // proj + window reverse + unshift + residual -> x2 (mode 1)
    gemm_wmma_kernel<<<dim3(MTOT/BM, CC/32), 256, 0, stream>>>(
        attnout, proj_wb, proj_b, nullptr, x2, x, MTOT, CC, CC, 1);

    // LN2 -> bf16
    ln_kernel<<<MTOT/8, 256, 0, stream>>>(x2, n2g, n2b, zbuf, 0);

    // FC1 + GELU (mode 2)
    gemm_wmma_kernel<<<dim3(MTOT/BM, HID/32), 256, 0, stream>>>(
        zbuf, fc1_wb, fc1_b, h1, nullptr, nullptr, MTOT, CC, HID, 2);

    // FC2 + residual -> d_out (mode 3)
    gemm_wmma_kernel<<<dim3(MTOT/BM, CC/32), 256, 0, stream>>>(
        h1, fc2_wb, fc2_b, nullptr, (float*)d_out, x2, MTOT, HID, CC, 3);

    (void)in_sizes; (void)n_in; (void)out_size; (void)ws_size;
}
